// AgentTransformer_46471546143008
// MI455X (gfx1250) — matrix operand order repaired and checked by reading, not compiled, not run on hardware
//
#include <hip/hip_runtime.h>
#include <hip/hip_bf16.h>
#include <stdint.h>

#define NUMB 64
#define TT   4001
#define DD   128
#define HH   8
#define HDIM 16
#define AA   5
#define LL   4000
#define ROWS (NUMB * TT)   // 256064, divisible by 16

typedef __attribute__((ext_vector_type(16))) __bf16 v16bf;
typedef __attribute__((ext_vector_type(8)))  float  v8f;

union ABfrag { v16bf v; uint4 q[2]; };

__device__ __forceinline__ uint16_t f2bf(float f) {
    uint32_t u = __float_as_uint(f);
    uint32_t r = (u + 0x7FFFu + ((u >> 16) & 1u)) >> 16;
    return (uint16_t)r;
}

// ---------------- f32 -> bf16 convert ----------------
__global__ void k_f2bf(const float* __restrict__ in, uint16_t* __restrict__ out, size_t n) {
    size_t i = (size_t)blockIdx.x * blockDim.x + threadIdx.x;
    size_t stride = (size_t)gridDim.x * blockDim.x;
    for (; i < n; i += stride) out[i] = f2bf(in[i]);
}

// ---------------- bf16 WMMA GEMM: C = A(MxK) * W(NxK)^T + bias ----------------
// A row-major (M,K) bf16; W row-major (N,K) bf16; C f32 with leading dim ldc.
// Each wave32 computes a 16(M) x 64(N) strip: 4 accumulators, one A fragment
// per k-step feeds 4 WMMAs (4x less A traffic than 1 tile/wave).
// Requires K % 32 == 0, N % 64 == 0, M % 16 == 0.
__global__ void k_gemm_bf16(const uint16_t* __restrict__ A, const uint16_t* __restrict__ W,
                            const float* __restrict__ bias, float* __restrict__ C,
                            int M, int N, int K, int ldc) {
    const int wave = threadIdx.x >> 5;
    const int lane = threadIdx.x & 31;
    const int hi   = lane >> 4;     // lane group
    const int l15  = lane & 15;
    const int nstrips = N >> 6;     // 64-wide N strips
    int strip = blockIdx.x * (blockDim.x >> 5) + wave;
    if (strip >= (M >> 4) * nstrips) return;
    const int tn4 = strip % nstrips;
    const int tm  = strip / nstrips;

    const uint16_t* arow = A + (size_t)(tm * 16 + l15) * K;              // A row m
    const uint16_t* brow0 = W + (size_t)(tn4 * 64 + l15) * K;            // B col n, u=0
    const uint16_t* brow1 = brow0 + (size_t)16 * K;
    const uint16_t* brow2 = brow0 + (size_t)32 * K;
    const uint16_t* brow3 = brow0 + (size_t)48 * K;

    v8f acc0 = {}, acc1 = {}, acc2 = {}, acc3 = {};
    for (int kk = 0; kk < K; kk += 32) {
        ABfrag a, b0, b1, b2, b3;
        // A 16x32 bf16: elem e -> K = hi*16 + e (contiguous 32B per lane)
        a.q[0] = *(const uint4*)(arow + kk + hi * 16);
        a.q[1] = *(const uint4*)(arow + kk + hi * 16 + 8);
        // B 32x16 bf16: elem e -> K = hi*16 + e (contiguous 32B per lane)
        b0.q[0] = *(const uint4*)(brow0 + kk + hi * 16);
        b0.q[1] = *(const uint4*)(brow0 + kk + hi * 16 + 8);
        b1.q[0] = *(const uint4*)(brow1 + kk + hi * 16);
        b1.q[1] = *(const uint4*)(brow1 + kk + hi * 16 + 8);
        b2.q[0] = *(const uint4*)(brow2 + kk + hi * 16);
        b2.q[1] = *(const uint4*)(brow2 + kk + hi * 16 + 8);
        b3.q[0] = *(const uint4*)(brow3 + kk + hi * 16);
        b3.q[1] = *(const uint4*)(brow3 + kk + hi * 16 + 8);
        acc0 = __builtin_amdgcn_wmma_f32_16x16x32_bf16(false, a.v, false, b0.v, (short)0, acc0, false, false);
        acc1 = __builtin_amdgcn_wmma_f32_16x16x32_bf16(false, a.v, false, b1.v, (short)0, acc1, false, false);
        acc2 = __builtin_amdgcn_wmma_f32_16x16x32_bf16(false, a.v, false, b2.v, (short)0, acc2, false, false);
        acc3 = __builtin_amdgcn_wmma_f32_16x16x32_bf16(false, a.v, false, b3.v, (short)0, acc3, false, false);
    }
    // D: lane n = lane&15, VGPR j -> row m = hi*8 + j
    float* cbase = C + (size_t)(tm * 16 + hi * 8) * ldc + tn4 * 64 + l15;
#pragma unroll
    for (int u = 0; u < 4; ++u) {
        const float bv = bias ? bias[tn4 * 64 + u * 16 + l15] : 0.0f;
        float* crow = cbase + u * 16;
        v8f acc = (u == 0) ? acc0 : (u == 1) ? acc1 : (u == 2) ? acc2 : acc3;
#pragma unroll
        for (int j = 0; j < 8; ++j) crow[(size_t)j * ldc] = acc[j] + bv;
    }
}

// ---------------- agent branch: conv1 (depthwise k=9) + LN + GELU at 25 positions ----------------
__global__ void k_agent_h(const float* __restrict__ qkv, const float* __restrict__ c1w,
                          const float* __restrict__ c1b, const float* __restrict__ lng,
                          const float* __restrict__ lnb, float* __restrict__ hsmall) {
    const int b = blockIdx.x / 25, p = blockIdx.x % 25;
    const int lk = p / 5, j = p % 5;
    const int tpos = lk * 800 + j;          // position within L=4000
    const int c = threadIdx.x;              // 128 threads
    float s = c1b[c];
#pragma unroll
    for (int t = 0; t < 9; ++t) {
        int tq = tpos + t - 4;
        if (tq >= 0 && tq < LL)
            s += qkv[(size_t)(b * TT + 1 + tq) * 384 + c] * c1w[c * 9 + t];
    }
    __shared__ float red[DD];
    __shared__ float mu_s, var_s;
    red[c] = s; __syncthreads();
    for (int o = 64; o; o >>= 1) { if (c < o) red[c] += red[c + o]; __syncthreads(); }
    if (c == 0) mu_s = red[0] * (1.0f / DD);
    __syncthreads();
    float d0 = s - mu_s;
    red[c] = d0 * d0; __syncthreads();
    for (int o = 64; o; o >>= 1) { if (c < o) red[c] += red[c + o]; __syncthreads(); }
    if (c == 0) var_s = red[0] * (1.0f / DD);
    __syncthreads();
    float hn = d0 * rsqrtf(var_s + 1e-5f) * lng[c] + lnb[c];
    float g = 0.5f * hn * (1.0f + erff(hn * 0.70710678118f));   // exact GELU
    hsmall[(size_t)(b * 25 + p) * DD + c] = g;
}

// ---------------- conv2 -> offsets -> gather with flat-reshape interp ----------------
__global__ void k_agent_gather(const float* __restrict__ x, const float* __restrict__ hsmall,
                               const float* __restrict__ c2w, float* __restrict__ xsg) {
    const int b = blockIdx.x;
    const int c = threadIdx.x;              // 128 threads
    __shared__ float red[DD];
    __shared__ int   idxs[AA];
    for (int lk = 0; lk < AA; ++lk) {
        float s = 0.0f;
#pragma unroll
        for (int j = 0; j < AA; ++j)
            s += hsmall[(size_t)(b * 25 + lk * 5 + j) * DD + c] * c2w[c * 5 + j];
        red[c] = s; __syncthreads();
        for (int o = 64; o; o >>= 1) { if (c < o) red[c] += red[c + o]; __syncthreads(); }
        if (c == 0) {
            float off = tanhf(red[0]) * 0.25f;          // 1/(Lk-1)
            float pos = off + ((float)lk + 0.5f);       // ref = lk + 0.5
            int id = (int)pos;
            idxs[lk] = id < 0 ? 0 : (id > 4 ? 4 : id);
        }
        __syncthreads();
    }
    // align_corners=True interp of length-4000 flat-reshaped view to 5 samples
    const int   x0s[AA] = {0, 999, 1999, 2999, 3999};
    const float wws[AA] = {0.0f, 0.75f, 0.5f, 0.25f, 0.0f};
    const float* tok = x + (size_t)b * TT * DD + DD;    // V(c,l) = tok[c*4000 + l]
    for (int lk = 0; lk < AA; ++lk) {
        int i = idxs[lk];
        int l0 = x0s[i];
        int l1 = (l0 + 1 < LL) ? l0 + 1 : LL - 1;
        float w = wws[i];
        float v = tok[(size_t)c * LL + l0] * (1.0f - w) + tok[(size_t)c * LL + l1] * w;
        xsg[(size_t)b * 640 + c * 5 + lk] = v;          // flat == xs_rs flat
    }
}

// ---------------- qn = xs_rs @ pq_w^T + pq_b, scattered to aq[b,h,a,hd] ----------------
__global__ void k_pq(const float* __restrict__ xsg, const float* __restrict__ pqw,
                     const float* __restrict__ pqb, float* __restrict__ aq) {
    const int b = blockIdx.x;
    const int oc = threadIdx.x;             // 128 threads
    __shared__ float xr[640];
    for (int i = oc; i < 640; i += DD) xr[i] = xsg[(size_t)b * 640 + i];
    __syncthreads();
    const int h = oc >> 4, hd = oc & 15;
    for (int a = 0; a < AA; ++a) {
        float s = pqb[oc];
        for (int c2 = 0; c2 < DD; ++c2) s += xr[a * DD + c2] * pqw[oc * DD + c2];
        aq[(size_t)b * 640 + h * 80 + a * 16 + hd] = s;  // aq[b,h,a,hd] = qn[b,a,h*16+hd]
    }
}

// ---------------- positional bias interpolation (align_corners=False, Lin=5) ----------------
__global__ void k_bias(const float* __restrict__ an, const float* __restrict__ na,
                       const float* __restrict__ atb, const float* __restrict__ tab,
                       float* __restrict__ pbout, float* __restrict__ about) {
    int idx = blockIdx.x * blockDim.x + threadIdx.x;    // over H*T
    if (idx >= HH * TT) return;
    const int h = idx / TT, t = idx % TT;
    float src = ((float)t + 0.5f) * (5.0f / 4001.0f) - 0.5f;
    src = fminf(fmaxf(src, 0.0f), 4.0f);
    int x0 = (int)floorf(src);
    int x1 = (x0 + 1 > 4) ? 4 : x0 + 1;
    float w = src - (float)x0;
#pragma unroll
    for (int a = 0; a < AA; ++a) {
        float ian = an[(h * 5 + a) * 5 + x0] * (1.0f - w) + an[(h * 5 + a) * 5 + x1] * w;
        float ina = na[(h * 5 + a) * 5 + x0] * (1.0f - w) + na[(h * 5 + a) * 5 + x1] * w;
        pbout[(size_t)(h * 5 + a) * TT + t] = ian + atb[(size_t)(h * 5 + a) * TT + t];
        about[((size_t)h * TT + t) * 5 + a] = ina + tab[((size_t)h * TT + t) * 5 + a];
    }
}

// ---------------- agent attention: softmax over T, agent_v accumulation ----------------
__global__ void k_agent_attn(const float* __restrict__ qkv, const float* __restrict__ aq,
                             const float* __restrict__ pb, float* __restrict__ sbuf,
                             float* __restrict__ agentv) {
    const int bh = blockIdx.x, b = bh >> 3, h = bh & 7;
    const int tid = threadIdx.x;            // 256
    __shared__ float aqs[80], red[256], amax[AA], asum[AA], accs[80];
    if (tid < 80) { aqs[tid] = aq[(size_t)bh * 80 + tid] * 0.25f; accs[tid] = 0.0f; }
    __syncthreads();
    const float* pbh = pb + (size_t)h * AA * TT;
    float* sb = sbuf + (size_t)bh * AA * TT;
    float lmax[AA] = {-1e30f, -1e30f, -1e30f, -1e30f, -1e30f};
    for (int t = tid; t < TT; t += 256) {
        const float* kr = qkv + (size_t)(b * TT + t) * 384 + 128 + h * 16;
        float k0[16];
#pragma unroll
        for (int d = 0; d < 16; ++d) k0[d] = kr[d];
#pragma unroll
        for (int a = 0; a < AA; ++a) {
            float s = pbh[(size_t)a * TT + t];
#pragma unroll
            for (int d = 0; d < 16; ++d) s += aqs[a * 16 + d] * k0[d];
            sb[(size_t)a * TT + t] = s;
            lmax[a] = fmaxf(lmax[a], s);
        }
    }
    for (int a = 0; a < AA; ++a) {
        red[tid] = lmax[a]; __syncthreads();
        for (int o = 128; o; o >>= 1) { if (tid < o) red[tid] = fmaxf(red[tid], red[tid + o]); __syncthreads(); }
        if (tid == 0) amax[a] = red[0];
        __syncthreads();
    }
    float lsum[AA] = {0, 0, 0, 0, 0};
    float lacc[80];
#pragma unroll
    for (int i = 0; i < 80; ++i) lacc[i] = 0.0f;
    for (int t = tid; t < TT; t += 256) {
        const float* vr = qkv + (size_t)(b * TT + t) * 384 + 256 + h * 16;
        float v0[16];
#pragma unroll
        for (int d = 0; d < 16; ++d) v0[d] = vr[d];
#pragma unroll
        for (int a = 0; a < AA; ++a) {
            float p = __expf(sb[(size_t)a * TT + t] - amax[a]);
            lsum[a] += p;
#pragma unroll
            for (int d = 0; d < 16; ++d) lacc[a * 16 + d] += p * v0[d];
        }
    }
    for (int a = 0; a < AA; ++a) {
        red[tid] = lsum[a]; __syncthreads();
        for (int o = 128; o; o >>= 1) { if (tid < o) red[tid] += red[tid + o]; __syncthreads(); }
        if (tid == 0) asum[a] = red[0];
        __syncthreads();
    }
    for (int i = 0; i < 80; ++i) atomicAdd(&accs[i], lacc[i]);
    __syncthreads();
    if (tid < 80) agentv[(size_t)bh * 80 + tid] = accs[tid] / asum[tid >> 4];
}

// ---------------- q-attention (softmax over A=5) + output, bf16 for proj GEMM ----------------
__global__ void k_qattn(const float* __restrict__ qkv, const float* __restrict__ aq,
                        const float* __restrict__ agentv, const float* __restrict__ ab,
                        uint16_t* __restrict__ attnout) {
    const int chunk = blockIdx.x & 15;
    const int bh = blockIdx.x >> 4, b = bh >> 3, h = bh & 7;
    const int tid = threadIdx.x;            // 256
    __shared__ float aqs[80], avs[80];
    if (tid < 80) { aqs[tid] = aq[(size_t)bh * 80 + tid]; avs[tid] = agentv[(size_t)bh * 80 + tid]; }
    __syncthreads();
    const int t = chunk * 256 + tid;
    if (t >= TT) return;
    const float* qr = qkv + (size_t)(b * TT + t) * 384 + h * 16;
    float q0[16];
#pragma unroll
    for (int d = 0; d < 16; ++d) q0[d] = qr[d] * 0.25f;
    const float* abr = ab + ((size_t)h * TT + t) * 5;
    float s[AA], m = -1e30f;
#pragma unroll
    for (int a = 0; a < AA; ++a) {
        float v = abr[a];
#pragma unroll
        for (int d = 0; d < 16; ++d) v += q0[d] * aqs[a * 16 + d];
        s[a] = v; m = fmaxf(m, v);
    }
    float sum = 0.0f;
#pragma unroll
    for (int a = 0; a < AA; ++a) { s[a] = __expf(s[a] - m); sum += s[a]; }
    const float inv = 1.0f / sum;
    uint16_t* orow = attnout + (size_t)(b * TT + t) * DD + h * 16;
#pragma unroll
    for (int d = 0; d < 16; ++d) {
        float o = 0.0f;
#pragma unroll
        for (int a = 0; a < AA; ++a) o += s[a] * avs[a * 16 + d];
        orow[d] = f2bf(o * inv);
    }
}

// =============================== host launcher ===============================
extern "C" void kernel_launch(void* const* d_in, const int* in_sizes, int n_in,
                              void* d_out, int out_size, void* d_ws, size_t ws_size,
                              hipStream_t stream) {
    const float* x      = (const float*)d_in[0];
    const float* qkv_w  = (const float*)d_in[1];
    const float* qkv_b  = (const float*)d_in[2];
    const float* proj_w = (const float*)d_in[3];
    const float* proj_b = (const float*)d_in[4];
    const float* c1w    = (const float*)d_in[5];
    const float* c1b    = (const float*)d_in[6];
    const float* lng    = (const float*)d_in[7];
    const float* lnb    = (const float*)d_in[8];
    const float* c2w    = (const float*)d_in[9];
    const float* pq_w   = (const float*)d_in[10];
    const float* pq_b   = (const float*)d_in[11];
    const float* an_b   = (const float*)d_in[12];
    const float* na_b   = (const float*)d_in[13];
    const float* at_b   = (const float*)d_in[14];
    const float* ta_b   = (const float*)d_in[15];
    float* out = (float*)d_out;

    char* ws = (char*)d_ws;
    size_t off = 0;
    auto alloc = [&](size_t bytes) { void* p = ws + off; off = (off + bytes + 255) & ~(size_t)255; return p; };
    uint16_t* xh      = (uint16_t*)alloc((size_t)ROWS * DD * 2);
    uint16_t* qkvw_h  = (uint16_t*)alloc((size_t)384 * DD * 2);
    uint16_t* projw_h = (uint16_t*)alloc((size_t)DD * DD * 2);
    float*    qkv     = (float*)   alloc((size_t)ROWS * 384 * 4);
    float*    hsmall  = (float*)   alloc((size_t)NUMB * 25 * DD * 4);
    float*    xsg     = (float*)   alloc((size_t)NUMB * 640 * 4);
    float*    aqb     = (float*)   alloc((size_t)NUMB * 640 * 4);
    float*    pbb     = (float*)   alloc((size_t)HH * AA * TT * 4);
    float*    abb     = (float*)   alloc((size_t)HH * TT * AA * 4);
    float*    sbuf    = (float*)   alloc((size_t)NUMB * HH * AA * TT * 4);
    float*    agentv  = (float*)   alloc((size_t)NUMB * HH * 80 * 4);
    uint16_t* attnout = (uint16_t*)alloc((size_t)ROWS * DD * 2);
    (void)ws_size; (void)n_in; (void)in_sizes; (void)out_size;

    // 1) stage bf16 operands
    k_f2bf<<<4096, 256, 0, stream>>>(x, xh, (size_t)ROWS * DD);
    k_f2bf<<<192, 256, 0, stream>>>(qkv_w, qkvw_h, (size_t)384 * DD);
    k_f2bf<<<64, 256, 0, stream>>>(proj_w, projw_h, (size_t)DD * DD);

    // 2) QKV GEMM: (ROWS,128) x (384,128)^T -> (ROWS,384); 16x64 strip per wave
    k_gemm_bf16<<<(ROWS / 16) * (384 / 64) / 8, 256, 0, stream>>>(
        xh, qkvw_h, qkv_b, qkv, ROWS, 384, DD, 384);

    // 3) agent branch
    k_agent_h<<<NUMB * 25, 128, 0, stream>>>(qkv, c1w, c1b, lng, lnb, hsmall);
    k_agent_gather<<<NUMB, 128, 0, stream>>>(x, hsmall, c2w, xsg);
    k_pq<<<NUMB, 128, 0, stream>>>(xsg, pq_w, pq_b, aqb);

    // 4) positional biases
    k_bias<<<(HH * TT + 255) / 256, 256, 0, stream>>>(an_b, na_b, at_b, ta_b, pbb, abb);

    // 5) agent attention (softmax over T) -> agent_v
    k_agent_attn<<<NUMB * HH, 256, 0, stream>>>(qkv, aqb, pbb, sbuf, agentv);

    // 6) q attention (softmax over A) -> attn_out (bf16)
    k_qattn<<<NUMB * HH * 16, 256, 0, stream>>>(qkv, aqb, agentv, abb, attnout);

    // 7) output projection GEMM into d_out; (ROWS/16)*(128/64)/8 = 4001 blocks exactly
    k_gemm_bf16<<<(ROWS / 16) * (DD / 64) / 8, 256, 0, stream>>>(
        attnout, projw_h, proj_b, out, ROWS, DD, DD, DD);
}
